// SP_DecoderLight_57982058496562
// MI455X (gfx1250) — matrix operand order repaired and checked by reading, not compiled, not run on hardware
//
#include <hip/hip_runtime.h>

// SP_DecoderLight on gfx1250 (MI455X): WMMA-f16 GEMM pipeline.
// Dead code eliminated: edge_block1 / AdaIN1 output is overwritten by
// conv_latent in the reference, so e1_* / ad1_* are never computed.
// All GEMMs: A [M,K] f16 row-major  x  W [N,K] f16 row-major (original
// weight layout == CDNA5 WMMA B-fragment-friendly), f32 accumulate.
// Workspace use ~620 MB (assumed available in d_ws).

typedef __attribute__((ext_vector_type(16))) _Float16 v16h;
typedef __attribute__((ext_vector_type(8)))  float    v8f;
typedef __attribute__((ext_vector_type(4)))  int      i4;
typedef __attribute__((ext_vector_type(2)))  _Float16 h2v;

#define B_  8
#define N_  2048
#define M1_ 16384      // B*N
#define KNB 10
#define MK_ 163840     // B*N*KNB

union FragU { v16h h; struct { i4 lo; i4 hi; } q; };

// A fragment (16x32 f16): lane<16 row=lane holds K{0..7,16..23}; lane>=16 K{8..15,24..31}
__device__ __forceinline__ v16h load_a_frag(const _Float16* p) {
  FragU f; f.q.lo = *(const i4*)p; f.q.hi = *(const i4*)(p + 16); return f.h;
}
// B fragment (32x16 f16): lane<16 col=lane holds K{0..7,16..23}; lane>=16 K{8..15,24..31}
__device__ __forceinline__ v16h load_b_frag(const _Float16* p) {
  FragU f; f.q.lo = *(const i4*)p; f.q.hi = *(const i4*)(p + 16); return f.h;
}

// EPI: 0=bias  1=bias+lrelu(slope)  2=acc+extra[row/10, col] (central term)
//      3=lrelu(acc+bias,slope) then AdaIN with extra=[row, 2N] (ga|be)
template <int EPI>
__global__ __launch_bounds__(256)
void gemm_f16_wmma(const _Float16* __restrict__ A, int lda,
                   const _Float16* __restrict__ Bw, int ldb,
                   const float* __restrict__ bias,
                   _Float16* __restrict__ C,
                   int M, int N, int K, float slope,
                   const _Float16* __restrict__ extra)
{
  const int wave = threadIdx.x >> 5;
  const int lane = threadIdx.x & 31;
  const int m0 = (blockIdx.x * 8 + wave) * 16;
  const int n0 = blockIdx.y * 64;
  const int hs = lane >> 4;       // half-wave select
  const int lr = lane & 15;

  const _Float16* pa = A  + (size_t)(m0 + lr) * lda + hs * 8;
  const _Float16* pb = Bw + (size_t)(n0 + lr) * ldb + hs * 8;

  v8f acc0 = {}, acc1 = {}, acc2 = {}, acc3 = {};
  for (int k = 0; k < K; k += 32) {
    v16h af = load_a_frag(pa + k);
    v16h b0 = load_b_frag(pb + k);
    v16h b1 = load_b_frag(pb + (size_t)16 * ldb + k);
    v16h b2 = load_b_frag(pb + (size_t)32 * ldb + k);
    v16h b3 = load_b_frag(pb + (size_t)48 * ldb + k);
    acc0 = __builtin_amdgcn_wmma_f32_16x16x32_f16(false, af, false, b0, (short)0, acc0, false, false);
    acc1 = __builtin_amdgcn_wmma_f32_16x16x32_f16(false, af, false, b1, (short)0, acc1, false, false);
    acc2 = __builtin_amdgcn_wmma_f32_16x16x32_f16(false, af, false, b2, (short)0, acc2, false, false);
    acc3 = __builtin_amdgcn_wmma_f32_16x16x32_f16(false, af, false, b3, (short)0, acc3, false, false);
  }

  const int rbase = m0 + hs * 8;   // C/D layout: lane<16 rows 0..7, lane>=16 rows 8..15
  #pragma unroll
  for (int j = 0; j < 4; j++) {
    v8f acc = (j == 0) ? acc0 : (j == 1) ? acc1 : (j == 2) ? acc2 : acc3;
    const int col = n0 + j * 16 + lr;
    const float bv = bias ? bias[col] : 0.0f;
    #pragma unroll
    for (int r = 0; r < 8; r++) {
      const int row = rbase + r;
      float v = acc[r] + bv;
      if (EPI == 1) v = (v >= 0.f) ? v : slope * v;
      if (EPI == 2) v += (float)extra[(size_t)(row / KNB) * N + col];
      if (EPI == 3) {
        v = (v >= 0.f) ? v : slope * v;
        const float ga = (float)extra[(size_t)row * (2 * N) + col];
        const float be = (float)extra[(size_t)row * (2 * N) + N + col];
        v = ga * v + be;
      }
      C[(size_t)row * N + col] = (_Float16)v;
    }
  }
}

// ---- packing / elementwise kernels ---------------------------------------

__global__ void pack_w_kernel(const float* __restrict__ src, _Float16* __restrict__ dst,
                              int rows, int src_ld, int dst_ld)
{
  int gid = blockIdx.x * 256 + threadIdx.x;
  if (gid >= rows * dst_ld) return;
  int r = gid / dst_ld, c = gid - r * dst_ld;
  dst[gid] = (c < src_ld) ? (_Float16)src[(size_t)r * src_ld + c] : (_Float16)0.f;
}

__global__ void pack_xcat_kernel(const float* __restrict__ x, const float* __restrict__ z,
                                 _Float16* __restrict__ dst)
{
  int gid = blockIdx.x * 256 + threadIdx.x;
  if (gid >= M1_ * 160) return;
  int bn = gid / 160, c = gid - bn * 160;
  int b = bn >> 11, n = bn & (N_ - 1);
  float v = 0.f;
  if (c < 3)        v = x[((size_t)b * 3 + c) * N_ + n];
  else if (c < 131) v = z[((size_t)b * 128 + (c - 3)) * N_ + n];
  dst[gid] = (_Float16)v;
}

// e2_ow [O=512, C=512, K=10] -> W'[o, kk] with kk = k*512 + c
__global__ void pack_ow_kernel(const float* __restrict__ src, _Float16* __restrict__ dst)
{
  size_t gid = (size_t)blockIdx.x * 256 + threadIdx.x;
  if (gid >= (size_t)512 * 5120) return;
  int o = (int)(gid / 5120), kk = (int)(gid - (size_t)o * 5120);
  int k = kk >> 9, c = kk & 511;
  dst[gid] = (_Float16)src[((size_t)o * 512 + c) * KNB + k];
}

__global__ __launch_bounds__(256)
void knn_kernel(const _Float16* __restrict__ X, int* __restrict__ idx)
{
  __shared__ h2v tile[64 * 64];                // 64 points x 128 halves = 16 KB
  int gid = blockIdx.x * 256 + threadIdx.x;    // one thread per (b, n)
  int b = gid >> 11, n = gid & (N_ - 1);
  const h2v* Xb = (const h2v*)X + (size_t)(b << 11) * 64;
  h2v xn[64];
  #pragma unroll
  for (int j = 0; j < 64; j++) xn[j] = Xb[(size_t)n * 64 + j];
  float bd[KNB]; int bi[KNB];
  #pragma unroll
  for (int k = 0; k < KNB; k++) { bd[k] = 3.0e38f; bi[k] = 0; }
  for (int mt = 0; mt < N_; mt += 64) {
    __syncthreads();
    for (int t = threadIdx.x; t < 64 * 64; t += 256)
      tile[t] = Xb[(size_t)(mt + (t >> 6)) * 64 + (t & 63)];
    __syncthreads();
    for (int mi = 0; mi < 64; mi++) {
      int m = mt + mi;
      if (m == n) continue;                    // exclude self (argsort[...,1:k+1])
      const h2v* xm = &tile[mi * 64];
      float dist = 0.f;
      #pragma unroll
      for (int j = 0; j < 64; j++) {
        float dx = (float)xn[j].x - (float)xm[j].x;
        float dy = (float)xn[j].y - (float)xm[j].y;
        dist += dx * dx + dy * dy;
      }
      // branch-free sorted insert, fully unrolled so bd/bi stay in VGPRs
      if (dist < bd[KNB - 1]) {
        #pragma unroll
        for (int p = KNB - 1; p >= 0; p--) {
          const float prev  = (p == 0) ? -1.0f : bd[p - 1];
          const int   previ = (p == 0) ? m     : bi[p - 1];
          if (dist < bd[p]) {
            const bool shift = (dist < prev);
            bd[p] = shift ? prev : dist;
            bi[p] = shift ? previ : m;
          }
        }
      }
    }
  }
  #pragma unroll
  for (int k = 0; k < KNB; k++) idx[(size_t)gid * KNB + k] = bi[k];
}

__global__ void diff_kernel(const _Float16* __restrict__ X, const int* __restrict__ idx,
                            _Float16* __restrict__ d)
{
  size_t gid = (size_t)blockIdx.x * 256 + threadIdx.x;
  if (gid >= (size_t)MK_ * 64) return;
  int r = (int)(gid >> 6);          // bnk
  int j = (int)(gid & 63);
  int bn = r / KNB;
  int b = bn >> 11;
  int m = idx[r];
  const h2v* Xv = (const h2v*)X;
  h2v nb = Xv[((size_t)(b << 11) + m) * 64 + j];
  h2v ce = Xv[(size_t)bn * 64 + j];
  ((h2v*)d)[gid] = nb - ce;
}

// ---- coalesced BatchNorm batch-stats: partial sums + finalize ------------
// Each wave covers 32 consecutive channels (64B/access), 8 waves stride rows,
// LDS-reduce across waves, then one f32 atomicAdd per channel per block.

__global__ void zero_kernel(float* __restrict__ p, int n)
{
  int i = blockIdx.x * 256 + threadIdx.x;
  if (i < n) p[i] = 0.f;
}

__global__ __launch_bounds__(256)
void stats_part_kernel(const _Float16* __restrict__ buf, int R, int C,
                       float* __restrict__ sum, float* __restrict__ sumsq)
{
  __shared__ float ps[8][32], ps2[8][32];
  const int lane = threadIdx.x & 31;
  const int wy   = threadIdx.x >> 5;
  const int c    = blockIdx.x * 32 + lane;
  const int chunk = R / gridDim.y;
  const int r0 = blockIdx.y * chunk;
  float s = 0.f, s2 = 0.f;
  for (int r = r0 + wy; r < r0 + chunk; r += 8) {
    float v = (float)buf[(size_t)r * C + c];
    s += v; s2 += v * v;
  }
  ps[wy][lane] = s; ps2[wy][lane] = s2;
  __syncthreads();
  if (wy == 0) {
    float ts = 0.f, ts2 = 0.f;
    #pragma unroll
    for (int j = 0; j < 8; j++) { ts += ps[j][lane]; ts2 += ps2[j][lane]; }
    atomicAdd(&sum[c], ts);
    atomicAdd(&sumsq[c], ts2);
  }
}

__global__ void stats_final_kernel(const float* __restrict__ sum, const float* __restrict__ sumsq,
                                   int R, int C, float* __restrict__ mean, float* __restrict__ rstd)
{
  int c = blockIdx.x * 256 + threadIdx.x;
  if (c >= C) return;
  float m = sum[c] / (float)R;
  float var = sumsq[c] / (float)R - m * m;
  mean[c] = m;
  rstd[c] = rsqrtf(var + 1e-5f);
}

__global__ void bn_lrelu_kernel(_Float16* __restrict__ buf, size_t total, int C,
                                const float* __restrict__ mean, const float* __restrict__ rstd,
                                const float* __restrict__ g, const float* __restrict__ b, float slope)
{
  size_t gid = (size_t)blockIdx.x * 256 + threadIdx.x;
  if (gid >= total) return;
  int c = (int)(gid & (size_t)(C - 1));       // C is a power of two (256/512)
  float v = (float)buf[gid];
  v = g[c] * (v - mean[c]) * rstd[c] + b[c];
  buf[gid] = (_Float16)((v >= 0.f) ? v : slope * v);
}

// softmax over k (axis=-1) of w, then h *= softmax(w); layout [bn, k, c]
__global__ void softmax_mul_kernel(const _Float16* __restrict__ w, _Float16* __restrict__ h)
{
  int gid = blockIdx.x * 256 + threadIdx.x;
  if (gid >= M1_ * 512) return;
  int bn = gid >> 9, c = gid & 511;
  size_t bse = ((size_t)bn * KNB) * 512 + c;
  float wv[KNB], mx = -3.0e38f;
  #pragma unroll
  for (int k = 0; k < KNB; k++) { wv[k] = (float)w[bse + (size_t)k * 512]; mx = fmaxf(mx, wv[k]); }
  float sum = 0.f;
  #pragma unroll
  for (int k = 0; k < KNB; k++) { wv[k] = expf(wv[k] - mx); sum += wv[k]; }
  float inv = 1.f / sum;
  #pragma unroll
  for (int k = 0; k < KNB; k++) {
    size_t p = bse + (size_t)k * 512;
    h[p] = (_Float16)((float)h[p] * wv[k] * inv);
  }
}

// final 64->3 conv + tanh, transposed store to [B,3,N] f32
__global__ void tail3_kernel(const _Float16* __restrict__ t2, const float* __restrict__ w3,
                             const float* __restrict__ b3, float* __restrict__ out)
{
  int gid = blockIdx.x * 256 + threadIdx.x;
  if (gid >= M1_) return;
  int b = gid >> 11, n = gid & (N_ - 1);
  float a0 = 0.f, a1 = 0.f, a2 = 0.f;
  #pragma unroll
  for (int j = 0; j < 64; j++) {
    float t = (float)t2[(size_t)gid * 64 + j];
    a0 += t * w3[j]; a1 += t * w3[64 + j]; a2 += t * w3[128 + j];
  }
  out[((size_t)b * 3 + 0) * N_ + n] = tanhf(a0 + b3[0]);
  out[((size_t)b * 3 + 1) * N_ + n] = tanhf(a1 + b3[1]);
  out[((size_t)b * 3 + 2) * N_ + n] = tanhf(a2 + b3[2]);
}

// ---- host ----------------------------------------------------------------

extern "C" void kernel_launch(void* const* d_in, const int* in_sizes, int n_in,
                              void* d_out, int out_size, void* d_ws, size_t ws_size,
                              hipStream_t stream)
{
  (void)in_sizes; (void)n_in; (void)out_size; (void)ws_size;
  const float* x       = (const float*)d_in[0];
  const float* z       = (const float*)d_in[1];
  const float* head_w1 = (const float*)d_in[2];
  const float* head_b1 = (const float*)d_in[3];
  const float* head_w2 = (const float*)d_in[4];
  const float* head_b2 = (const float*)d_in[5];
  const float* lat_w   = (const float*)d_in[22];
  const float* lat_b   = (const float*)d_in[23];
  const float* e2_w1   = (const float*)d_in[24];
  const float* e2_b1   = (const float*)d_in[25];
  const float* e2_g1   = (const float*)d_in[26];
  const float* e2_bb1  = (const float*)d_in[27];
  const float* e2_w2   = (const float*)d_in[28];
  const float* e2_b2   = (const float*)d_in[29];
  const float* e2_g2   = (const float*)d_in[30];
  const float* e2_bb2  = (const float*)d_in[31];
  const float* e2_xw   = (const float*)d_in[32];
  const float* e2_xb   = (const float*)d_in[33];
  const float* e2_xg   = (const float*)d_in[34];
  const float* e2_xbb  = (const float*)d_in[35];
  const float* e2_ow   = (const float*)d_in[36];
  const float* e2_ob   = (const float*)d_in[37];
  const float* ad2_w   = (const float*)d_in[38];
  const float* ad2_b   = (const float*)d_in[39];
  const float* tail_w1 = (const float*)d_in[40];
  const float* tail_b1 = (const float*)d_in[41];
  const float* tail_w2 = (const float*)d_in[42];
  const float* tail_b2 = (const float*)d_in[43];
  const float* tail_w3 = (const float*)d_in[44];
  const float* tail_b3 = (const float*)d_in[45];
  float* out = (float*)d_out;

  char* base = (char*)d_ws;
  size_t off = 0;
  auto alloc = [&](size_t bytes) -> char* {
    char* p = base + off; off += (bytes + 255) & ~(size_t)255; return p;
  };
  auto ah = [&](size_t e) -> _Float16* { return (_Float16*)alloc(e * sizeof(_Float16)); };
  auto af = [&](size_t e) -> float*    { return (float*)alloc(e * sizeof(float)); };
  auto ai = [&](size_t e) -> int*      { return (int*)alloc(e * sizeof(int)); };

  _Float16* xcat   = ah((size_t)M1_ * 160);
  _Float16* w_h1   = ah((size_t)512 * 160);
  _Float16* w_h2   = ah((size_t)512 * 512);
  _Float16* w_lat  = ah((size_t)128 * 512);
  _Float16* w_ad2  = ah((size_t)1024 * 512);
  _Float16* w_e2w1 = ah((size_t)256 * 128);
  _Float16* w_e2w2 = ah((size_t)512 * 256);
  _Float16* w_xw   = ah((size_t)512 * 256);
  _Float16* w_ow   = ah((size_t)512 * 5120);
  _Float16* w_t1   = ah((size_t)256 * 512);
  _Float16* w_t2   = ah((size_t)64 * 256);
  _Float16* style1 = ah((size_t)M1_ * 512);
  _Float16* style  = ah((size_t)M1_ * 512);
  _Float16* x1     = ah((size_t)M1_ * 128);
  int*      nidx   = ai((size_t)M1_ * KNB);
  _Float16* diffb  = ah((size_t)MK_ * 128);
  _Float16* c1     = ah((size_t)MK_ * 256);
  _Float16* c2     = ah((size_t)MK_ * 512);
  _Float16* hbuf   = ah((size_t)MK_ * 512);
  _Float16* hc     = ah((size_t)M1_ * 512);
  _Float16* gabe   = ah((size_t)M1_ * 1024);
  _Float16* x2     = ah((size_t)M1_ * 512);
  _Float16* t1     = ah((size_t)M1_ * 256);
  _Float16* t2     = ah((size_t)M1_ * 64);
  float*    sumb   = af(512);
  float*    sumsqb = af(512);
  float*    meanb  = af(512);
  float*    rstdb  = af(512);

  auto grid1 = [](size_t total) { return dim3((unsigned)((total + 255) / 256)); };

  auto gemm = [&](int epi, const _Float16* A, int lda, const _Float16* Bw, int ldb,
                  const float* bias, _Float16* Cc, int M, int N, int K, float slope,
                  const _Float16* extra) {
    dim3 g(M / 128, N / 64), blk(256);
    switch (epi) {
      case 0: gemm_f16_wmma<0><<<g, blk, 0, stream>>>(A, lda, Bw, ldb, bias, Cc, M, N, K, slope, extra); break;
      case 1: gemm_f16_wmma<1><<<g, blk, 0, stream>>>(A, lda, Bw, ldb, bias, Cc, M, N, K, slope, extra); break;
      case 2: gemm_f16_wmma<2><<<g, blk, 0, stream>>>(A, lda, Bw, ldb, bias, Cc, M, N, K, slope, extra); break;
      default: gemm_f16_wmma<3><<<g, blk, 0, stream>>>(A, lda, Bw, ldb, bias, Cc, M, N, K, slope, extra); break;
    }
  };

  auto stats = [&](const _Float16* buf, int R, int C) {
    zero_kernel<<<2, 256, 0, stream>>>(sumb, 512);
    zero_kernel<<<2, 256, 0, stream>>>(sumsqb, 512);
    dim3 g(C / 32, 64);
    stats_part_kernel<<<g, 256, 0, stream>>>(buf, R, C, sumb, sumsqb);
    stats_final_kernel<<<grid1(C), 256, 0, stream>>>(sumb, sumsqb, R, C, meanb, rstdb);
  };

  // ---- pack inputs/weights to f16 (K padded to multiples of 32) ----
  pack_xcat_kernel<<<grid1((size_t)M1_ * 160), 256, 0, stream>>>(x, z, xcat);
  pack_w_kernel<<<grid1((size_t)512 * 160), 256, 0, stream>>>(head_w1, w_h1, 512, 131, 160);
  pack_w_kernel<<<grid1((size_t)512 * 512), 256, 0, stream>>>(head_w2, w_h2, 512, 512, 512);
  pack_w_kernel<<<grid1((size_t)128 * 512), 256, 0, stream>>>(lat_w,   w_lat, 128, 512, 512);
  pack_w_kernel<<<grid1((size_t)1024 * 512), 256, 0, stream>>>(ad2_w,  w_ad2, 1024, 512, 512);
  pack_w_kernel<<<grid1((size_t)256 * 128), 256, 0, stream>>>(e2_w1,  w_e2w1, 256, 128, 128);
  pack_w_kernel<<<grid1((size_t)512 * 256), 256, 0, stream>>>(e2_w2,  w_e2w2, 512, 256, 256);
  pack_w_kernel<<<grid1((size_t)512 * 256), 256, 0, stream>>>(e2_xw,  w_xw,   512, 256, 256);
  pack_w_kernel<<<grid1((size_t)256 * 512), 256, 0, stream>>>(tail_w1, w_t1,  256, 512, 512);
  pack_w_kernel<<<grid1((size_t)64 * 256), 256, 0, stream>>>(tail_w2, w_t2,   64, 256, 256);
  pack_ow_kernel<<<grid1((size_t)512 * 5120), 256, 0, stream>>>(e2_ow, w_ow);

  // ---- style MLP ----
  gemm(1, xcat,   160, w_h1, 160, head_b1, style1, M1_, 512, 160, 0.01f, nullptr);
  gemm(1, style1, 512, w_h2, 512, head_b2, style,  M1_, 512, 512, 0.01f, nullptr);

  // ---- x1 = conv_latent(style) (edge_block1 result is dead) ----
  gemm(0, style, 512, w_lat, 512, lat_b, x1, M1_, 128, 512, 0.f, nullptr);

  // ---- kNN + edge features (diff part) ----
  knn_kernel<<<M1_ / 256, 256, 0, stream>>>(x1, nidx);
  diff_kernel<<<grid1((size_t)MK_ * 64), 256, 0, stream>>>(x1, nidx, diffb);

  // ---- edge_block2 weight branch: conv -> BN(batch stats) -> lrelu, twice ----
  gemm(0, diffb, 128, w_e2w1, 128, e2_b1, c1, MK_, 256, 128, 0.f, nullptr);
  stats(c1, MK_, 256);
  bn_lrelu_kernel<<<grid1((size_t)MK_ * 256), 256, 0, stream>>>(c1, (size_t)MK_ * 256, 256, meanb, rstdb, e2_g1, e2_bb1, 0.01f);
  gemm(0, c1, 256, w_e2w2, 256, e2_b2, c2, MK_, 512, 256, 0.f, nullptr);
  stats(c2, MK_, 512);
  bn_lrelu_kernel<<<grid1((size_t)MK_ * 512), 256, 0, stream>>>(c2, (size_t)MK_ * 512, 512, meanb, rstdb, e2_g2, e2_bb2, 0.01f);

  // ---- x branch: conv_x(ee) = Wc*x1 (per-point) + Wd*diff (per-edge) ----
  gemm(0, x1,    128, w_xw,       256, e2_xb,  hc,   M1_, 512, 128, 0.f, nullptr);
  gemm(2, diffb, 128, w_xw + 128, 256, nullptr, hbuf, MK_, 512, 128, 0.f, hc);
  stats(hbuf, MK_, 512);
  bn_lrelu_kernel<<<grid1((size_t)MK_ * 512), 256, 0, stream>>>(hbuf, (size_t)MK_ * 512, 512, meanb, rstdb, e2_xg, e2_xbb, 0.01f);

  // ---- h *= softmax_k(w) ----
  softmax_mul_kernel<<<grid1((size_t)M1_ * 512), 256, 0, stream>>>(c2, hbuf);

  // ---- AdaIN2 params, then conv_out (K=5120) fused with lrelu(0.2)+AdaIN ----
  gemm(0, style, 512, w_ad2, 512, ad2_b, gabe, M1_, 1024, 512, 0.f, nullptr);
  gemm(3, hbuf, 5120, w_ow, 5120, e2_ob, x2, M1_, 512, 5120, 0.2f, gabe);

  // ---- tail MLP + tanh ----
  gemm(1, x2, 512, w_t1, 512, tail_b1, t1, M1_, 256, 512, 0.01f, nullptr);
  gemm(1, t1, 256, w_t2, 256, tail_b2, t2, M1_, 64, 256, 0.01f, nullptr);
  tail3_kernel<<<grid1(M1_), 256, 0, stream>>>(t2, tail_w3, tail_b3, out);
}
